// Block_46059229282655
// MI455X (gfx1250) — matrix operand order repaired and checked by reading, not compiled, not run on hardware
//
#include <hip/hip_runtime.h>

// ---------------------------------------------------------------------------
// Types / helpers
// ---------------------------------------------------------------------------
typedef __attribute__((ext_vector_type(16))) __bf16 v16bf;
typedef __attribute__((ext_vector_type(8)))  float  v8f;

union FragAB {
  v16bf v;
  __bf16 h[16];
  uint4  q[2];
};

__device__ __forceinline__ float gelu_f(float x) {
  // exact erf GELU (torch nn.GELU default)
  return 0.5f * x * (1.0f + erff(x * 0.70710678118654752f));
}

// Generic-pointer -> LDS byte offset (for async-to-LDS inline asm operands)
__device__ __forceinline__ unsigned lds_off(const void* p) {
  return (unsigned)(unsigned long long)(const __attribute__((address_space(3))) void*)p;
}

// ---------------------------------------------------------------------------
// Weight prep kernels: f32 (R x C) -> bf16 transposed (C x R), and straight cvt
// ---------------------------------------------------------------------------
__global__ __launch_bounds__(256) void transpose_cvt(const float* __restrict__ in,
                                                     __bf16* __restrict__ out,
                                                     int R, int C) {
  __shared__ float t[32][33];
  int c0 = blockIdx.x * 32, r0 = blockIdx.y * 32;
  int tx = threadIdx.x & 31, ty = threadIdx.x >> 5; // 32x8
#pragma unroll
  for (int i = 0; i < 4; ++i)
    t[ty + i * 8][tx] = in[(size_t)(r0 + ty + i * 8) * C + c0 + tx];
  __syncthreads();
#pragma unroll
  for (int i = 0; i < 4; ++i)
    out[(size_t)(c0 + ty + i * 8) * R + r0 + tx] = (__bf16)t[tx][ty + i * 8];
}

__global__ __launch_bounds__(256) void cvt_bf16(const float* __restrict__ in,
                                                __bf16* __restrict__ out, int n) {
  int i = blockIdx.x * 256 + threadIdx.x;
  if (i < n) out[i] = (__bf16)in[i];
}

// ---------------------------------------------------------------------------
// LayerNorm over C=768, one row per block, writes bf16 (feeds a GEMM)
// ---------------------------------------------------------------------------
__global__ __launch_bounds__(256) void ln_kernel(const float* __restrict__ x,
                                                 const float* __restrict__ g,
                                                 const float* __restrict__ b,
                                                 __bf16* __restrict__ out) {
  const int C = 768;
  int row = blockIdx.x, tid = threadIdx.x;
  const float* xr = x + (size_t)row * C;
  float v[3];
#pragma unroll
  for (int i = 0; i < 3; ++i) v[i] = xr[tid + i * 256];
  float s = v[0] + v[1] + v[2];
  float sq = v[0] * v[0] + v[1] * v[1] + v[2] * v[2];
  __shared__ float r1[256], r2[256];
  r1[tid] = s; r2[tid] = sq;
  __syncthreads();
  for (int off = 128; off > 0; off >>= 1) {
    if (tid < off) { r1[tid] += r1[tid + off]; r2[tid] += r2[tid + off]; }
    __syncthreads();
  }
  float mean = r1[0] * (1.0f / C);
  float var  = r2[0] * (1.0f / C) - mean * mean;
  float rstd = rsqrtf(var + 1e-5f);
#pragma unroll
  for (int i = 0; i < 3; ++i) {
    int c = tid + i * 256;
    out[(size_t)row * C + c] = (__bf16)((v[i] - mean) * rstd * g[c] + b[c]);
  }
}

// ---------------------------------------------------------------------------
// Tiled bf16 WMMA GEMM:  C[M x Nn] = A[M x K] * Bt[Nn x K]^T  (+ epilogue)
// Block: 256 threads = 8 waves (4x2), tile 128x64, BK=64.
// Data movement: double-buffered GLOBAL_LOAD_ASYNC_TO_LDS_B128 (ASYNCcnt).
// The pipeline is unrolled by two tiles so both LDS buffers are statically
// addressed (no cndmask/branches in the steady-state loop).
// Each wave: 32x32 => 2x2 tiles of v_wmma_f32_16x16x32_bf16.
// Epilogue: optional bias, residual(f32), BN(eval), exact GELU; f32/bf16 out.
// ---------------------------------------------------------------------------
#define GBM 128
#define GBN 64
#define GBK 64

__global__ __launch_bounds__(256) void gemm_bf16(
    const __bf16* __restrict__ A, const __bf16* __restrict__ Bt,
    const float* __restrict__ bias, const float* __restrict__ res,
    const float* __restrict__ bng, const float* __restrict__ bnb,
    const float* __restrict__ bnm, const float* __restrict__ bnv,
    float* __restrict__ outF, __bf16* __restrict__ outB,
    int M, int Nn, int K, int ldc, int coff, int doGelu) {
  (void)M;
  __shared__ __attribute__((aligned(16))) __bf16 As[2][GBM][GBK + 8];
  __shared__ __attribute__((aligned(16))) __bf16 Bs[2][GBN][GBK + 8];

  int tid = threadIdx.x;
  int lane = tid & 31, wid = tid >> 5;
  int wm = wid >> 1, wn = wid & 1;
  int l15 = lane & 15, hi = lane >> 4;
  int m0 = blockIdx.y * GBM, n0 = blockIdx.x * GBN;

  v8f acc[2][2] = {};

  // per-thread tile slices: A 64B (4x b128), B 32B (2x b128)
  int aRow = tid >> 1, aHalf = (tid & 1) * 32;
  int bRow = tid >> 2, bChunk = (tid & 3) * 16;

  unsigned la0 = lds_off(&As[0][aRow][aHalf]);
  unsigned la1 = lds_off(&As[1][aRow][aHalf]);
  unsigned lb0 = lds_off(&Bs[0][bRow][bChunk]);
  unsigned lb1 = lds_off(&Bs[1][bRow][bChunk]);
  unsigned long long ga =
      (unsigned long long)(A + (size_t)(m0 + aRow) * K + aHalf);
  unsigned long long gb =
      (unsigned long long)(Bt + (size_t)(n0 + bRow) * K + bChunk);

  auto issue = [&](unsigned la, unsigned lb) {
    asm volatile("global_load_async_to_lds_b128 %0, %1, off"
                 :: "v"(la), "v"(ga) : "memory");
    asm volatile("global_load_async_to_lds_b128 %0, %1, off offset:16"
                 :: "v"(la), "v"(ga) : "memory");
    asm volatile("global_load_async_to_lds_b128 %0, %1, off offset:32"
                 :: "v"(la), "v"(ga) : "memory");
    asm volatile("global_load_async_to_lds_b128 %0, %1, off offset:48"
                 :: "v"(la), "v"(ga) : "memory");
    asm volatile("global_load_async_to_lds_b128 %0, %1, off"
                 :: "v"(lb), "v"(gb) : "memory");
    asm volatile("global_load_async_to_lds_b128 %0, %1, off offset:16"
                 :: "v"(lb), "v"(gb) : "memory");
    ga += (unsigned long long)(GBK * 2);
    gb += (unsigned long long)(GBK * 2);
  };

  auto compute = [&](int buf) {
#pragma unroll
    for (int ks = 0; ks < 2; ++ks) {
      FragAB a[2], b[2];
#pragma unroll
      for (int i = 0; i < 2; ++i) {
        int row = wm * 32 + i * 16 + l15;
        int kb = ks * 32 + (hi ? 8 : 0);
        a[i].q[0] = *(const uint4*)&As[buf][row][kb];
        a[i].q[1] = *(const uint4*)&As[buf][row][kb + 16];
      }
#pragma unroll
      for (int j = 0; j < 2; ++j) {
        int col = wn * 32 + j * 16 + l15;
        int kb = ks * 32 + hi * 8;
        const uint4* p = (const uint4*)&Bs[buf][col][kb];
        b[j].q[0] = p[0]; b[j].q[1] = p[2];
      }
#pragma unroll
      for (int i = 0; i < 2; ++i)
#pragma unroll
        for (int j = 0; j < 2; ++j)
          acc[i][j] = __builtin_amdgcn_wmma_f32_16x16x32_bf16(
              false, a[i].v, false, b[j].v, (short)0, acc[i][j], false, false);
    }
  };

  // Software pipeline, T = K/GBK tiles (always even here: 12 or 48).
  // in-order async completion: wait asynccnt<=6 => previous tile's 6 ops done.
  const int T = K / GBK;
  issue(la0, lb0); // tile 0 -> buf0
  for (int t = 0; t + 2 < T; t += 2) {
    issue(la1, lb1);
    asm volatile("s_wait_asynccnt 6" ::: "memory");
    __syncthreads();
    compute(0);
    __syncthreads();
    issue(la0, lb0);
    asm volatile("s_wait_asynccnt 6" ::: "memory");
    __syncthreads();
    compute(1);
    __syncthreads();
  }
  // final pair of tiles
  issue(la1, lb1);
  asm volatile("s_wait_asynccnt 6" ::: "memory");
  __syncthreads();
  compute(0);
  __syncthreads();
  asm volatile("s_wait_asynccnt 0" ::: "memory");
  __syncthreads();
  compute(1);

  // epilogue
#pragma unroll
  for (int i = 0; i < 2; ++i)
#pragma unroll
    for (int j = 0; j < 2; ++j) {
#pragma unroll
      for (int r = 0; r < 8; ++r) {
        int gM = m0 + wm * 32 + i * 16 + hi * 8 + r;
        int gN = n0 + wn * 32 + j * 16 + l15;
        float vv = acc[i][j][r];
        if (bias) vv += bias[gN];
        if (res)  vv += res[(size_t)gM * Nn + gN];
        if (bng)  vv = (vv - bnm[gN]) * rsqrtf(bnv[gN] + 1e-5f) * bng[gN] + bnb[gN];
        if (doGelu) vv = gelu_f(vv);
        size_t oi = (size_t)gM * ldc + coff + gN;
        if (outF) outF[oi] = vv;
        if (outB) outB[oi] = (__bf16)vv;
      }
    }
}

// ---------------------------------------------------------------------------
// Flash attention (non-causal), bf16 WMMA, f32 accumulation.
// Q/K/V are bf16 columns inside a row-major buffer with stride ld (qkv layout):
//   Q[n*ld + head*HD + d], K at base +768, V at base +1536 (cols).
// Fragments load directly as contiguous b128 pairs (wave32 A/B layouts);
// softmax scale is applied to S (cheaper than per-element Q scaling).
// Block: 256 thr = 8 waves, 128 q-rows per block (16/wave), key tile 64.
// ---------------------------------------------------------------------------
template <int HD>
__global__ __launch_bounds__(256) void attn_kernel(
    const __bf16* __restrict__ Q, const __bf16* __restrict__ Kp,
    const __bf16* __restrict__ Vp, int ld, __bf16* __restrict__ O, int ldo,
    int N, float scale) {
  constexpr int KT = 64;
  __shared__ __attribute__((aligned(16))) __bf16 Vt[HD][KT + 8];
  __shared__ __attribute__((aligned(16))) __bf16 Pl[8][16][KT + 8];

  int tid = threadIdx.x, lane = tid & 31, wid = tid >> 5;
  int l15 = lane & 15, hi = lane >> 4;
  int head = blockIdx.y, hoff = head * HD;
  int q0 = blockIdx.x * 128 + wid * 16;

  // Q fragments (A layout), loaded once, zero-conversion b128 loads
  FragAB qf[HD / 32];
#pragma unroll
  for (int kk = 0; kk < HD / 32; ++kk) {
    int row = q0 + l15;
    int kb = kk * 32 + (hi ? 8 : 0);
    const __bf16* p = Q + (size_t)row * ld + hoff + kb;
    qf[kk].q[0] = *(const uint4*)p;
    qf[kk].q[1] = *(const uint4*)(p + 16);
  }

  v8f o[HD / 16] = {};
  float mrow[8], lrow[8];
#pragma unroll
  for (int r = 0; r < 8; ++r) { mrow[r] = -3.0e38f; lrow[r] = 0.0f; }

  for (int kt = 0; kt < N; kt += KT) {
    __syncthreads(); // previous Vt consumers done
    // cooperative V tile load (bf16 b128) transposed into Vt[hd][key]
    for (int chunk = tid; chunk < KT * (HD / 8); chunk += 256) {
      int key = chunk / (HD / 8);
      int c = (chunk % (HD / 8)) * 8;
      union { uint4 u; __bf16 e[8]; } vv;
      vv.u = *(const uint4*)(Vp + (size_t)(kt + key) * ld + hoff + c);
#pragma unroll
      for (int e = 0; e < 8; ++e) Vt[c + e][key] = vv.e[e];
    }

    // S = Q K^T for this key tile: 4 column tiles of 16 keys
    v8f s[4] = {};
#pragma unroll
    for (int kk = 0; kk < HD / 32; ++kk) {
#pragma unroll
      for (int j = 0; j < 4; ++j) {
        FragAB b;
        const __bf16* p =
            Kp + (size_t)(kt + j * 16 + l15) * ld + hoff + kk * 32 + hi * 8;
        b.q[0] = *(const uint4*)p;
        b.q[1] = *(const uint4*)(p + 16);
        s[j] = __builtin_amdgcn_wmma_f32_16x16x32_bf16(
            false, qf[kk].v, false, b.v, (short)0, s[j], false, false);
      }
    }
#pragma unroll
    for (int j = 0; j < 4; ++j)
#pragma unroll
      for (int r = 0; r < 8; ++r) s[j][r] *= scale;

    // online softmax (row = hi*8 + r, cols across lanes 0..15 x 4 tiles)
    float alpha[8];
#pragma unroll
    for (int r = 0; r < 8; ++r) {
      float t = fmaxf(fmaxf(s[0][r], s[1][r]), fmaxf(s[2][r], s[3][r]));
#pragma unroll
      for (int m = 8; m >= 1; m >>= 1) t = fmaxf(t, __shfl_xor(t, m, 32));
      float mn = fmaxf(mrow[r], t);
      alpha[r] = __expf(mrow[r] - mn);
      mrow[r] = mn;
    }
    float rs[8];
#pragma unroll
    for (int r = 0; r < 8; ++r) rs[r] = 0.0f;
#pragma unroll
    for (int j = 0; j < 4; ++j) {
#pragma unroll
      for (int r = 0; r < 8; ++r) {
        float pv = __expf(s[j][r] - mrow[r]);
        rs[r] += pv;
        Pl[wid][hi * 8 + r][j * 16 + l15] = (__bf16)pv;
      }
    }
#pragma unroll
    for (int r = 0; r < 8; ++r) {
      float t = rs[r];
#pragma unroll
      for (int m = 8; m >= 1; m >>= 1) t += __shfl_xor(t, m, 32);
      lrow[r] = lrow[r] * alpha[r] + t;
    }
#pragma unroll
    for (int n = 0; n < HD / 16; ++n)
#pragma unroll
      for (int r = 0; r < 8; ++r) o[n][r] *= alpha[r];

    __syncthreads(); // Vt ready for all waves

    // O += P V  (P re-read as A-fragments from per-wave LDS)
#pragma unroll
    for (int kk = 0; kk < 2; ++kk) {
      FragAB pf;
      int kb = kk * 32 + (hi ? 8 : 0);
      pf.q[0] = *(const uint4*)&Pl[wid][l15][kb];
      pf.q[1] = *(const uint4*)&Pl[wid][l15][kb + 16];
#pragma unroll
      for (int n = 0; n < HD / 16; ++n) {
        FragAB vf;
        int c = n * 16 + l15;
        int kb2 = kk * 32 + hi * 8;
        const uint4* pv = (const uint4*)&Vt[c][kb2];
        vf.q[0] = pv[0]; vf.q[1] = pv[2];
        o[n] = __builtin_amdgcn_wmma_f32_16x16x32_bf16(
            false, pf.v, false, vf.v, (short)0, o[n], false, false);
      }
    }
  }

  // normalize and write bf16 output (row-major N x (nh*HD))
#pragma unroll
  for (int r = 0; r < 8; ++r) {
    float inv = 1.0f / lrow[r];
    int grow = q0 + hi * 8 + r;
#pragma unroll
    for (int n = 0; n < HD / 16; ++n)
      O[(size_t)grow * ldo + hoff + n * 16 + l15] = (__bf16)(o[n][r] * inv);
  }
}

// ---------------------------------------------------------------------------
// Host orchestration
// ---------------------------------------------------------------------------
extern "C" void kernel_launch(void* const* d_in, const int* in_sizes, int n_in,
                              void* d_out, int out_size, void* d_ws, size_t ws_size,
                              hipStream_t stream) {
  (void)in_sizes; (void)n_in; (void)out_size; (void)ws_size;
  const int N = 4096, C = 768, MLP = 3072;

  const float* x      = (const float*)d_in[0];
  const float* y      = (const float*)d_in[1];
  const float* z      = (const float*)d_in[2];
  const float* ln1_g  = (const float*)d_in[3];
  const float* ln1_b  = (const float*)d_in[4];
  const float* ln2_g  = (const float*)d_in[5];
  const float* ln2_b  = (const float*)d_in[6];
  const float* qkv_w  = (const float*)d_in[7];
  const float* qkv_b  = (const float*)d_in[8];
  const float* proj_w = (const float*)d_in[9];
  const float* proj_b = (const float*)d_in[10];
  const float* caq_w  = (const float*)d_in[11];
  const float* cak_w  = (const float*)d_in[12];
  const float* cav_w  = (const float*)d_in[13];
  const float* cao_w  = (const float*)d_in[14];
  const float* cao_b  = (const float*)d_in[15];
  const float* mw1    = (const float*)d_in[16];
  const float* mb1    = (const float*)d_in[17];
  const float* mw2    = (const float*)d_in[18];
  const float* mb2    = (const float*)d_in[19];
  const float* m2w1   = (const float*)d_in[20];
  const float* m2b1   = (const float*)d_in[21];
  const float* m2w2   = (const float*)d_in[22];
  const float* m2b2   = (const float*)d_in[23];
  const float* pw1_w  = (const float*)d_in[24];
  const float* pw1_g  = (const float*)d_in[25];
  const float* pw1_bb = (const float*)d_in[26];
  const float* pw1_m  = (const float*)d_in[27];
  const float* pw1_v  = (const float*)d_in[28];
  const float* pw2_w  = (const float*)d_in[29];
  const float* pw2_g  = (const float*)d_in[30];
  const float* pw2_bb = (const float*)d_in[31];
  const float* pw2_m  = (const float*)d_in[32];
  const float* pw2_v  = (const float*)d_in[33];

  float* out = (float*)d_out;
  float* prompt1 = out;                   // N x C
  float* out_pw1 = out + (size_t)N * C;   // N x C
  float* out_pw2 = out + (size_t)2 * N * C;

  // workspace carve-out
  char* ws = (char*)d_ws;
  size_t off = 0;
  auto alloc = [&](size_t bytes) -> void* {
    void* p = ws + off;
    off += (bytes + 255) & ~(size_t)255;
    return p;
  };

  __bf16* qkvT  = (__bf16*)alloc((size_t)3 * C * C * 2 * 2); // 2304x768 bf16
  __bf16* projT = (__bf16*)alloc((size_t)C * C * 2);
  __bf16* caqT  = (__bf16*)alloc((size_t)C * C * 2);
  __bf16* cakT  = (__bf16*)alloc((size_t)C * C * 2);
  __bf16* cavT  = (__bf16*)alloc((size_t)C * C * 2);
  __bf16* caoT  = (__bf16*)alloc((size_t)C * C * 2);
  __bf16* m2w1T = (__bf16*)alloc((size_t)MLP * C * 2);
  __bf16* m2w2T = (__bf16*)alloc((size_t)C * MLP * 2);
  __bf16* mw1T  = (__bf16*)alloc((size_t)MLP * C * 2);
  __bf16* mw2T  = (__bf16*)alloc((size_t)C * MLP * 2);
  __bf16* pw1B  = (__bf16*)alloc((size_t)C * C * 2);
  __bf16* pw2B  = (__bf16*)alloc((size_t)C * C * 2);

  __bf16* qkvB  = (__bf16*)alloc((size_t)N * 3 * C * 2);  // N x 2304 bf16
  float*  x1a   = (float*)alloc((size_t)N * C * 4);       // x1_pre / x2
  __bf16* hA    = (__bf16*)alloc((size_t)N * C * 2);      // LN outputs (reused)
  __bf16* hidB  = (__bf16*)alloc((size_t)N * MLP * 2);    // MLP hidden bf16
  __bf16* aoutB = (__bf16*)alloc((size_t)N * C * 2);      // attn out bf16
  __bf16* p2B   = (__bf16*)alloc((size_t)N * C * 2);      // p2 bf16

  // ---- weight prep ----
  struct TW { const float* s; __bf16* d; int R, Cc; };
  TW tw[10] = {
      {qkv_w, qkvT, C, 3 * C}, {proj_w, projT, C, C},
      {caq_w, caqT, C, C},     {cak_w, cakT, C, C},
      {cav_w, cavT, C, C},     {cao_w, caoT, C, C},
      {m2w1, m2w1T, C, MLP},   {m2w2, m2w2T, MLP, C},
      {mw1, mw1T, C, MLP},     {mw2, mw2T, MLP, C}};
  for (int i = 0; i < 10; ++i)
    transpose_cvt<<<dim3(tw[i].Cc / 32, tw[i].R / 32), 256, 0, stream>>>(
        tw[i].s, tw[i].d, tw[i].R, tw[i].Cc);
  cvt_bf16<<<(C * C + 255) / 256, 256, 0, stream>>>(pw1_w, pw1B, C * C);
  cvt_bf16<<<(C * C + 255) / 256, 256, 0, stream>>>(pw2_w, pw2B, C * C);

  auto gemm = [&](const __bf16* A, const __bf16* Bt, const float* bias,
                  const float* res, const float* bg, const float* bb,
                  const float* bm, const float* bv, float* oF, __bf16* oB,
                  int M, int Nn, int K, int ldc, int coff, int gelu) {
    gemm_bf16<<<dim3(Nn / GBN, M / GBM), 256, 0, stream>>>(
        A, Bt, bias, res, bg, bb, bm, bv, oF, oB, M, Nn, K, ldc, coff, gelu);
  };

  // ---- stage 1: self-attention ----
  ln_kernel<<<N, 256, 0, stream>>>(x, ln1_g, ln1_b, hA);
  gemm(hA, qkvT, qkv_b, nullptr, nullptr, nullptr, nullptr, nullptr,
       nullptr, qkvB, N, 3 * C, C, 3 * C, 0, 0);
  attn_kernel<64><<<dim3(N / 128, 12), 256, 0, stream>>>(
      qkvB, qkvB + C, qkvB + 2 * C, 3 * C, aoutB, C, N, 0.125f);
  gemm(aoutB, projT, proj_b, x, nullptr, nullptr, nullptr, nullptr,
       x1a, nullptr, N, C, C, C, 0, 0);

  // MLP2 (uses ln2)
  ln_kernel<<<N, 256, 0, stream>>>(x1a, ln2_g, ln2_b, hA);
  gemm(hA, m2w1T, m2b1, nullptr, nullptr, nullptr, nullptr, nullptr,
       nullptr, hidB, N, MLP, C, MLP, 0, 1);
  gemm(hidB, m2w2T, m2b2, x1a, nullptr, nullptr, nullptr, nullptr,
       prompt1, nullptr, N, C, MLP, C, 0, 0);   // x1 final = prompt1

  // ---- stage 2: cross-attention (q from x1, k from y, v from z) ----
  ln_kernel<<<N, 256, 0, stream>>>(prompt1, ln1_g, ln1_b, hA);
  gemm(hA, caqT, nullptr, nullptr, nullptr, nullptr, nullptr, nullptr,
       nullptr, qkvB, N, C, C, 3 * C, 0, 0);
  ln_kernel<<<N, 256, 0, stream>>>(y, ln1_g, ln1_b, hA);
  gemm(hA, cakT, nullptr, nullptr, nullptr, nullptr, nullptr, nullptr,
       nullptr, qkvB, N, C, C, 3 * C, C, 0);
  ln_kernel<<<N, 256, 0, stream>>>(z, ln1_g, ln1_b, hA);
  gemm(hA, cavT, nullptr, nullptr, nullptr, nullptr, nullptr, nullptr,
       nullptr, qkvB, N, C, C, 3 * C, 2 * C, 0);
  attn_kernel<96><<<dim3(N / 128, 8), 256, 0, stream>>>(
      qkvB, qkvB + C, qkvB + 2 * C, 3 * C, aoutB, C, N, 0.10206207261596575f);
  gemm(aoutB, caoT, cao_b, prompt1, nullptr, nullptr, nullptr, nullptr,
       x1a, nullptr, N, C, C, C, 0, 0);          // x1a now holds x2

  // MLP (uses ln2)
  ln_kernel<<<N, 256, 0, stream>>>(x1a, ln2_g, ln2_b, hA);
  gemm(hA, mw1T, mb1, nullptr, nullptr, nullptr, nullptr, nullptr,
       nullptr, hidB, N, MLP, C, MLP, 0, 1);
  gemm(hidB, mw2T, mb2, x1a, nullptr, nullptr, nullptr, nullptr,
       nullptr, p2B, N, C, MLP, C, 0, 0);        // p2 (bf16 only)

  // ---- pointwise conv heads: 1x1 conv -> BN(eval) -> GELU ----
  gemm(p2B, pw1B, nullptr, nullptr, pw1_g, pw1_bb, pw1_m, pw1_v,
       out_pw1, nullptr, N, C, C, C, 0, 1);
  gemm(p2B, pw2B, nullptr, nullptr, pw2_g, pw2_bb, pw2_m, pw2_v,
       out_pw2, nullptr, N, C, C, C, 0, 1);
}
